// AdaMoLE_76845554860268
// MI455X (gfx1250) — matrix operand order repaired and checked by reading, not compiled, not run on hardware
//
#include <hip/hip_runtime.h>
#include <hip/hip_bf16.h>

// ---------------------------------------------------------------------------
// AdaMoLE fused:  OUT = ((relu(softmax(XR)+b - thr) normalized) * (X A_e^T)) B_e * 2
// Two bf16 WMMA GEMMs + fp32 gating epilogue.
//   GEMM1: [8192,4096] x [4096,144]  (128 LoRA-A cols + 8 router + 1 thr + 7 pad)
//   GEMM2: [8192,128]  x [128,4096]
// Memory-bound: ~274 MB HBM => ~11.8us at 23.3 TB/s. Intermediates L2-resident.
// This rev: sched_group_barrier directives (guarded by __has_builtin) pin the
// DS-read / WMMA interleave so B-fragment loads stay batched ahead of their
// WMMAs -> distinct registers + partial s_wait_dscnt instead of full drains.
// ---------------------------------------------------------------------------

typedef __attribute__((ext_vector_type(16))) __bf16 bf16x16;
typedef __attribute__((ext_vector_type(8)))  float  f32x8;

#define DIN   4096
#define DOUT  4096
#define NTOK  8192

// Scheduling-group directive (no-op if builtin unavailable).
// masks: 0x8 = MFMA/WMMA group, 0x100 = DS-read group
#if defined(__has_builtin)
#if __has_builtin(__builtin_amdgcn_sched_group_barrier)
#define SGB(mask, cnt, id) __builtin_amdgcn_sched_group_barrier((mask), (cnt), (id))
#endif
#endif
#ifndef SGB
#define SGB(mask, cnt, id)
#endif

// fp32 -> bf16 round-to-nearest-even via bit ops
static __device__ __forceinline__ unsigned short f2bf(float f) {
  unsigned u = __builtin_bit_cast(unsigned, f);
  unsigned r = (u + 0x7FFFu + ((u >> 16) & 1u)) >> 16;
  return (unsigned short)r;
}

// ---------------------------------------------------------------------------
// Prep: Wcat (4096 x 144) in WMMA B-fragment order, bf16.
//   col c<128: lora_A flat row c; 128-135: router_w; 136: thr_w; >136: 0 pad.
// Block (ct, ks) = 1024B: element (lane L, i) -> k = ks*32 + i + 16*(L>=16),
// c = ct*16 + (L&15).  Linear idx = ((ct*128+ks)*512) + L*16 + i.
// ---------------------------------------------------------------------------
__global__ __launch_bounds__(256) void prep_w_kernel(
    const float* __restrict__ lora_A, const float* __restrict__ router_w,
    const float* __restrict__ thr_w, unsigned short* __restrict__ WcatB) {
  int idx = blockIdx.x * 256 + threadIdx.x;           // < 9*128*512 = 589824
  int i  = idx & 15;
  int L  = (idx >> 4) & 31;
  int ks = (idx >> 9) & 127;
  int ct = idx >> 16;
  int k  = ks * 32 + i + ((L & 16) ? 16 : 0);
  int c  = ct * 16 + (L & 15);
  float v;
  if (c < 128)       v = lora_A[c * DIN + k];
  else if (c < 136)  v = router_w[(c - 128) * DIN + k];
  else if (c == 136) v = thr_w[k];
  else               v = 0.0f;
  WcatB[idx] = f2bf(v);
}

// ---------------------------------------------------------------------------
// Prep: Bcat (128 x 4096) in WMMA B-fragment order, bf16.
//   Bcat[k=e*16+r][o] = lora_B[e][o][r]   (lora_B is [8,4096,16])
// ---------------------------------------------------------------------------
__global__ __launch_bounds__(256) void prep_b_kernel(
    const float* __restrict__ lora_B, unsigned short* __restrict__ BcatB) {
  int idx = blockIdx.x * 256 + threadIdx.x;           // < 256*4*512 = 524288
  int i  = idx & 15;
  int L  = (idx >> 4) & 31;
  int ks = (idx >> 9) & 3;
  int ot = idx >> 11;
  int k  = ks * 32 + i + ((L & 16) ? 16 : 0);
  int o  = ot * 16 + (L & 15);
  int e = k >> 4, r = k & 15;
  BcatB[idx] = f2bf(lora_B[(e * DOUT + o) * 16 + r]);
}

// ---------------------------------------------------------------------------
// Kernel 1: GEMM1 + gating epilogue.
//   256 WGs x 64 threads (2 waves). WG: 32 rows x 144 cols, K=4096.
//   Each wave owns one 16-row tile and all 9 column tiles.
// ---------------------------------------------------------------------------
__global__ __launch_bounds__(64) void k1_gemm_gate_kernel(
    const float* __restrict__ X, const float* __restrict__ router_b,
    const float* __restrict__ thr_b, const unsigned short* __restrict__ WcatB,
    unsigned short* __restrict__ WHfrag) {
  // LDS: A bufs [2][2KB] at 0/2048, B bufs [2][9KB] at 4096/13312 (total 22528).
  // Epilogue reuses: per-wave 5632B region (gate 1KB, w 512B, h 4KB) x 2 waves.
  __shared__ alignas(32) unsigned char smem[22528];

  const int tid  = threadIdx.x;
  const int w    = tid >> 5;        // wave 0..1 = row tile
  const int lane = tid & 31;
  const int col  = lane & 15;
  const int hi   = lane >> 4;
  const int row0 = blockIdx.x * 32;

  f32x8 acc[9] = {};

  auto stageA = [&](int it, int buf) {
    const float4* Xv = reinterpret_cast<const float4*>(X + (size_t)row0 * DIN + it * 32);
    unsigned short* A = reinterpret_cast<unsigned short*>(smem + buf * 2048);
    for (int idx = tid; idx < 256; idx += 64) {
      int r = idx >> 3, kq = idx & 7;
      float4 v = Xv[r * 1024 + kq];
      if (it + 2 < 128) __builtin_prefetch(&Xv[r * 1024 + kq + 16], 0, 0);
      unsigned short b0 = f2bf(v.x), b1 = f2bf(v.y), b2 = f2bf(v.z), b3 = f2bf(v.w);
      int rt = r >> 4;
      int ln = (r & 15) | ((kq >> 2) << 4);
      int i0 = (kq & 3) << 2;
      uint2 pk = make_uint2((unsigned)b0 | ((unsigned)b1 << 16),
                            (unsigned)b2 | ((unsigned)b3 << 16));
      *reinterpret_cast<uint2*>(A + rt * 512 + ln * 16 + i0) = pk;
    }
  };
  auto stageB = [&](int it, int buf) {
    const uint4* src = reinterpret_cast<const uint4*>(WcatB) + it * 64;
    uint4* dst = reinterpret_cast<uint4*>(smem + 4096 + buf * 9216);
    for (int c = tid; c < 576; c += 64) {
      int ct = c >> 6, q = c & 63;
      dst[c] = src[ct * 8192 + q];       // block (ct,it) at (ct*128+it)*64 uint4
    }
  };

  stageA(0, 0);
  stageB(0, 0);
  for (int it = 0; it < 128; ++it) {
    __syncthreads();
    if (it + 1 < 128) { stageA(it + 1, (it + 1) & 1); stageB(it + 1, (it + 1) & 1); }
    const int cb = it & 1;
    const unsigned short* A = reinterpret_cast<const unsigned short*>(smem + cb * 2048);
    const unsigned short* Bb = reinterpret_cast<const unsigned short*>(smem + 4096 + cb * 9216);

    bf16x16 af = *reinterpret_cast<const bf16x16*>(A + w * 512 + lane * 16);
#define LOADB(ct) (*reinterpret_cast<const bf16x16*>(Bb + (ct) * 512 + lane * 16))
#define WMMA(a, b, c) __builtin_amdgcn_wmma_f32_16x16x32_bf16(false, (a), false, (b), (short)0, (c), false, false)
    // explicit 3-deep pipeline: each B-frag load issued 3 WMMAs before its use
    bf16x16 bb0 = LOADB(0);
    bf16x16 bb1 = LOADB(1);
    bf16x16 bb2 = LOADB(2);
    acc[0] = WMMA(af, bb0, acc[0]);  bb0 = LOADB(3);
    acc[1] = WMMA(af, bb1, acc[1]);  bb1 = LOADB(4);
    acc[2] = WMMA(af, bb2, acc[2]);  bb2 = LOADB(5);
    acc[3] = WMMA(af, bb0, acc[3]);  bb0 = LOADB(6);
    acc[4] = WMMA(af, bb1, acc[4]);  bb1 = LOADB(7);
    acc[5] = WMMA(af, bb2, acc[5]);  bb2 = LOADB(8);
    acc[6] = WMMA(af, bb0, acc[6]);
    acc[7] = WMMA(af, bb1, acc[7]);
    acc[8] = WMMA(af, bb2, acc[8]);
#undef LOADB
    // Pin the pipeline shape: 8 DS-reads (A-frag + b0..b2), then
    // 6 x [1 WMMA, 2 DS-reads], then the last 3 WMMAs.
    SGB(0x100, 8, 0);
#pragma unroll
    for (int g = 0; g < 6; ++g) { SGB(0x8, 1, 0); SGB(0x100, 2, 0); }
    SGB(0x8, 3, 0);
  }
  __syncthreads();

  // ---- gating epilogue (fp32) ----
  unsigned char* ep = smem + w * 5632;
  float* gate_f        = reinterpret_cast<float*>(ep);                // 16x16 f32
  float* w_f           = reinterpret_cast<float*>(ep + 1024);         // 16x8  f32
  unsigned short* h16  = reinterpret_cast<unsigned short*>(ep + 1536);// 16x128 bf16

  // dump column-tile 8 (cols 128..143: 8 logits, 1 thr-logit, 7 zero pad)
#pragma unroll
  for (int j = 0; j < 8; ++j) gate_f[(j + 8 * hi) * 16 + col] = acc[8][j];
  __syncthreads();

  if (lane < 16) {
    int r = lane;
    float l[9];
#pragma unroll
    for (int e = 0; e < 9; ++e) l[e] = gate_f[r * 16 + e];
    float m = -3.4e38f;
#pragma unroll
    for (int e = 0; e < 8; ++e) { l[e] += router_b[e]; m = fmaxf(m, l[e]); }
    float p[8], s = 0.0f;
#pragma unroll
    for (int e = 0; e < 8; ++e) { p[e] = __expf(l[e] - m); s += p[e]; }
    float inv = 1.0f / s;
    float thr = 0.125f / (1.0f + __expf(-(l[8] + thr_b[0])));   // sigmoid * 1/E
    float ws = 0.0f;
#pragma unroll
    for (int e = 0; e < 8; ++e) {
      float g = p[e] * inv - thr;     // adapted; relu == mask*adapted
      g = g > 0.0f ? g : 0.0f;
      p[e] = g; ws += g;
    }
    ws = (ws == 0.0f) ? 1.0f : ws;
    float sc = 2.0f / ws;             // fold SCALING = alpha/r = 2.0
#pragma unroll
    for (int e = 0; e < 8; ++e) w_f[r * 8 + e] = p[e] * sc;
  }
  __syncthreads();

  // apply weights; H -> bf16 row-major scratch (col tile ct == expert e)
#pragma unroll
  for (int ct = 0; ct < 8; ++ct) {
#pragma unroll
    for (int j = 0; j < 8; ++j) {
      int r = j + 8 * hi;
      float wg = w_f[r * 8 + ct];
      h16[r * 128 + ct * 16 + col] = f2bf(acc[ct][j] * wg);
    }
  }
  __syncthreads();

  // write WH in A-fragment order: block (nt, ks) 1024B, lane L gets 32B:
  // i 0..15 -> k = ks*32 + base + i  (base = 16*hi)
  {
    int nt = blockIdx.x * 2 + w;
    int ra = col;                     // A-frag: lane holds row L%16
    int base = hi ? 16 : 0;
#pragma unroll
    for (int ks = 0; ks < 4; ++ks) {
      uint4 c0 = *reinterpret_cast<const uint4*>(h16 + ra * 128 + ks * 32 + base);
      uint4 c1 = *reinterpret_cast<const uint4*>(h16 + ra * 128 + ks * 32 + base + 8);
      uint4* d = reinterpret_cast<uint4*>(
          reinterpret_cast<unsigned char*>(WHfrag) + (size_t)(nt * 4 + ks) * 1024 + lane * 32);
      d[0] = c0; d[1] = c1;
    }
  }
}

// ---------------------------------------------------------------------------
// Kernel 2: OUT = WH @ Bcat  ([8192,128] x [128,4096], fp32 out).
//   2048 WGs x 256 threads (8 waves). WG = 8 row-tiles x 8 o-tiles.
//   A-frags in registers (4 k-steps); B tile (4KB) staged in LDS shared by WG.
// ---------------------------------------------------------------------------
__global__ __launch_bounds__(256) void k2_gemm_out_kernel(
    const unsigned short* __restrict__ WHfrag,
    const unsigned short* __restrict__ BcatB, float* __restrict__ out) {
  __shared__ alignas(32) unsigned char smem[4096];
  const int tid  = threadIdx.x;
  const int w    = tid >> 5;
  const int lane = tid & 31;
  const int col  = lane & 15;
  const int hi   = lane >> 4;
  const int ng   = blockIdx.x >> 5;   // 64 row groups of 8 tiles
  const int og   = blockIdx.x & 31;   // 32 o groups of 8 tiles
  const int nt   = ng * 8 + w;

  bf16x16 aF[4];
#pragma unroll
  for (int ks = 0; ks < 4; ++ks)
    aF[ks] = *reinterpret_cast<const bf16x16*>(
        reinterpret_cast<const unsigned char*>(WHfrag) +
        (size_t)(nt * 4 + ks) * 1024 + lane * 32);

  for (int t = 0; t < 8; ++t) {
    int ot = og * 8 + t;
    __syncthreads();                  // previous tile's LDS reads done
    reinterpret_cast<uint4*>(smem)[tid] =
        reinterpret_cast<const uint4*>(BcatB)[ot * 256 + tid];
    __syncthreads();

    bf16x16 b0 = *reinterpret_cast<const bf16x16*>(smem + 0 * 1024 + lane * 32);
    bf16x16 b1 = *reinterpret_cast<const bf16x16*>(smem + 1 * 1024 + lane * 32);
    bf16x16 b2 = *reinterpret_cast<const bf16x16*>(smem + 2 * 1024 + lane * 32);
    bf16x16 b3 = *reinterpret_cast<const bf16x16*>(smem + 3 * 1024 + lane * 32);
    f32x8 acc = {};
    acc = WMMA(aF[0], b0, acc);
    acc = WMMA(aF[1], b1, acc);
    acc = WMMA(aF[2], b2, acc);
    acc = WMMA(aF[3], b3, acc);
    // batch all 8 DS-reads ahead of the 4 chained WMMAs
    SGB(0x100, 8, 0);
    SGB(0x8, 4, 0);

    float* orow = out + (size_t)(nt * 16) * DOUT + ot * 16 + col;
#pragma unroll
    for (int j = 0; j < 8; ++j)
      orow[(size_t)(j + 8 * hi) * DOUT] = acc[j];
  }
}
#undef WMMA

// ---------------------------------------------------------------------------
extern "C" void kernel_launch(void* const* d_in, const int* in_sizes, int n_in,
                              void* d_out, int out_size, void* d_ws, size_t ws_size,
                              hipStream_t stream) {
  (void)in_sizes; (void)n_in; (void)out_size; (void)ws_size;
  const float* X        = (const float*)d_in[0];
  const float* router_w = (const float*)d_in[1];
  const float* router_b = (const float*)d_in[2];
  const float* thr_w    = (const float*)d_in[3];
  const float* thr_b    = (const float*)d_in[4];
  const float* lora_A   = (const float*)d_in[5];
  const float* lora_B   = (const float*)d_in[6];
  float* out = (float*)d_out;

  unsigned char* ws = (unsigned char*)d_ws;
  unsigned short* WcatB  = (unsigned short*)(ws);                       // 1,179,648 B
  unsigned short* BcatB  = (unsigned short*)(ws + 1179648);             // 1,048,576 B
  unsigned short* WHfrag = (unsigned short*)(ws + 1179648 + 1048576);   // 2,097,152 B

  prep_w_kernel<<<2304, 256, 0, stream>>>(lora_A, router_w, thr_w, WcatB);
  prep_b_kernel<<<2048, 256, 0, stream>>>(lora_B, BcatB);
  k1_gemm_gate_kernel<<<256, 64, 0, stream>>>(X, router_b, thr_b, WcatB, WHfrag);
  k2_gemm_out_kernel<<<2048, 256, 0, stream>>>(WHfrag, BcatB, out);
}
